// RelativeAttention_10067403341880
// MI455X (gfx1250) — matrix operand order repaired and checked by reading, not compiled, not run on hardware
//
#include <hip/hip_runtime.h>
#include <hip/hip_bf16.h>

typedef __bf16 bf16_t;
typedef __bf16 v16bf __attribute__((ext_vector_type(16)));
typedef float  v8f   __attribute__((ext_vector_type(8)));
typedef int    v4i_  __attribute__((vector_size(16)));   // matches builtin param

typedef __attribute__((address_space(1))) v4i_* gv4i_p;  // global int4*
typedef __attribute__((address_space(3))) v4i_* lv4i_p;  // LDS int4*

union FragBF {
    v16bf v;
    uint4 u[2];
};

#define B_     8
#define N_     1024
#define H_     8
#define DH_    64
#define DIM_   512
#define INNER_ 512
#define QKV_LD 1536   // 3*INNER

#define LOG2E_ 1.4426950408889634f

// ---------------------------------------------------------------------------
// CDNA5 async global->LDS copy (ASYNCcnt-tracked); sync fallback if the
// builtin is unavailable on this toolchain.
// ---------------------------------------------------------------------------
__device__ __forceinline__ void async_copy_b128(const bf16_t* g, bf16_t* l) {
#if __has_builtin(__builtin_amdgcn_global_load_async_to_lds_b128)
    __builtin_amdgcn_global_load_async_to_lds_b128(
        (gv4i_p)(bf16_t*)g, (lv4i_p)l, 0, 0);
#else
    *(uint4*)l = *(const uint4*)g;
#endif
}

__device__ __forceinline__ void async_wait0() {
#if __has_builtin(__builtin_amdgcn_s_wait_asynccnt)
    __builtin_amdgcn_s_wait_asynccnt(0);
#endif
}

__device__ __forceinline__ float fast_exp2(float x) {
#if __has_builtin(__builtin_amdgcn_exp2f)
    return __builtin_amdgcn_exp2f(x);
#else
    return exp2f(x);
#endif
}

// ---------------------------------------------------------------------------
// fp32 -> bf16 conversion
// ---------------------------------------------------------------------------
__global__ void cvt_f32_bf16(const float* __restrict__ src,
                             bf16_t* __restrict__ dst, int n) {
    int i = blockIdx.x * blockDim.x + threadIdx.x;
    if (i < n) dst[i] = (bf16_t)src[i];
}

// ---------------------------------------------------------------------------
// bf16 WMMA GEMM: C[M,N] = A[M,K] @ B[K,N] (+bias). A,B row-major bf16.
// 128x128 tile per 256-thread block (8 waves). K-step 32, LDS double-buffered:
// next A tile staged via async-to-LDS, next B tile fetched to regs during the
// current tile's WMMAs; one barrier per k-step. Fragment loads are all 16B
// ds_load_b128 matching the 16-bit WMMA VGPR layouts.
// ---------------------------------------------------------------------------
template <bool OUT_BF16, bool HAS_BIAS>
__global__ __launch_bounds__(256)
void gemm_bf16(const bf16_t* __restrict__ A, const bf16_t* __restrict__ Bm,
               const float* __restrict__ bias, void* __restrict__ Cout,
               int M, int N, int K) {
    __shared__ __align__(16) bf16_t sA[2][128 * 40];
    __shared__ __align__(16) bf16_t sB[2][128 * 40];  // [n][k]

    const int t    = threadIdx.x;
    const int lane = t & 31;
    const int wid  = t >> 5;
    const int l15  = lane & 15;
    const int lh   = lane >> 4;
    const int wm   = (wid >> 1) * 32;
    const int wn   = (wid & 1) * 64;
    const int m0   = blockIdx.y * 128;
    const int n0   = blockIdx.x * 128;

    v8f acc[2][4];
    const v8f vzero = {0.f, 0.f, 0.f, 0.f, 0.f, 0.f, 0.f, 0.f};
#pragma unroll
    for (int mf = 0; mf < 2; ++mf)
#pragma unroll
        for (int nf = 0; nf < 4; ++nf) acc[mf][nf] = vzero;

    const int arow = t >> 1, aseg = (t & 1) * 16;   // A: 2 thr/row, 16 bf16 each
    const int brow = t >> 3, bseg = (t & 7) * 16;   // B: 8 thr/row, 16 bf16 each
    const bf16_t* aptr  = A + (size_t)(m0 + arow) * K + aseg;     // + k
    const bf16_t* bptr0 = Bm + (size_t)brow * N + n0 + bseg;      // + k*N

    const int KT = K >> 5;

    union BStage { uint4 u[2]; bf16_t h[16]; } bst;

    // ---- prologue: stage k-step 0 ----
    async_copy_b128(aptr,     &sA[0][arow * 40 + aseg]);
    async_copy_b128(aptr + 8, &sA[0][arow * 40 + aseg + 8]);
    {
        const uint4* bs = (const uint4*)bptr0;
        bst.u[0] = bs[0]; bst.u[1] = bs[1];
    }
#pragma unroll
    for (int j = 0; j < 16; ++j) sB[0][(bseg + j) * 40 + brow] = bst.h[j];

    for (int kt = 0; kt < KT; ++kt) {
        async_wait0();        // our async writes into sA[cur] are done
        __syncthreads();      // everyone's writes visible; prev readers done
        const int cur = kt & 1, nxt = cur ^ 1;
        const bool have_next = (kt + 1) < KT;
        if (have_next) {
            const int k1 = (kt + 1) * 32;
            // async-stage next A tile; readers of sA[nxt] passed the barrier
            async_copy_b128(aptr + k1,     &sA[nxt][arow * 40 + aseg]);
            async_copy_b128(aptr + k1 + 8, &sA[nxt][arow * 40 + aseg + 8]);
            const uint4* bs = (const uint4*)(bptr0 + (size_t)k1 * N);
            bst.u[0] = bs[0]; bst.u[1] = bs[1];
        }

#pragma unroll
        for (int mf = 0; mf < 2; ++mf) {
            FragBF a;  // A 16x32: lane half h holds K = {h*8.., 16+h*8..}
            const bf16_t* ap = &sA[cur][(wm + mf * 16 + l15) * 40 + lh * 8];
            a.u[0] = *(const uint4*)(ap);
            a.u[1] = *(const uint4*)(ap + 16);
#pragma unroll
            for (int nf = 0; nf < 4; ++nf) {
                FragBF b;  // B 32x16: lane half h holds K = {h*8.., 16+h*8..}
                const bf16_t* bp = &sB[cur][(wn + nf * 16 + l15) * 40 + lh * 8];
                b.u[0] = *(const uint4*)(bp);
                b.u[1] = *(const uint4*)(bp + 16);
                acc[mf][nf] = __builtin_amdgcn_wmma_f32_16x16x32_bf16(
                    false, a.v, false, b.v, (short)0, acc[mf][nf], false, false);
            }
        }

        if (have_next) {  // transpose-store next B tile into the other buffer
#pragma unroll
            for (int j = 0; j < 16; ++j) sB[nxt][(bseg + j) * 40 + brow] = bst.h[j];
        }
    }

    // --- writeback: C frag row = r + 8*lh, col = l15 ---
#pragma unroll
    for (int mf = 0; mf < 2; ++mf)
#pragma unroll
        for (int nf = 0; nf < 4; ++nf) {
            const int row = m0 + wm + mf * 16 + lh * 8;
            const int col = n0 + wn + nf * 16 + l15;
#pragma unroll
            for (int r = 0; r < 8; ++r) {
                if (OUT_BF16) {
                    ((bf16_t*)Cout)[(size_t)(row + r) * N + col] = (bf16_t)acc[mf][nf][r];
                } else {
                    float v = acc[mf][nf][r];
                    if (HAS_BIAS) v += bias[col];
                    ((float*)Cout)[(size_t)(row + r) * N + col] = v;
                }
            }
        }
}

// ---------------------------------------------------------------------------
// Flash attention with relative position bias.
// Block = 256 threads (8 waves); block handles one (b,h) and 128 query rows
// (wave owns 16). Loops over 16 key-chunks of 64 with online softmax in the
// log2 domain (v_exp_f32 is natively exp2). K staged via async-to-LDS; V
// staged with coalesced b128 loads + b16 transpose stores into LDS.
// ---------------------------------------------------------------------------
__global__ __launch_bounds__(256)
void flash_attn_relbias(const bf16_t* __restrict__ qkv,
                        const float* __restrict__ bias_table,
                        bf16_t* __restrict__ attn_out) {
    __shared__ __align__(16) bf16_t sQ[128 * 72];  // [row][d]
    __shared__ __align__(16) bf16_t sK[64 * 72];   // [key][d]  == B-frag layout for Q@K^T
    __shared__ __align__(16) bf16_t sVt[64 * 72];  // [d][key]  == B-frag layout for P@V
    __shared__ __align__(16) bf16_t sP[128 * 72];  // [row][key], per-wave private rows
    __shared__ float sBias[192];                   // log2e * bias, by diagonal i-j+63

    const int t    = threadIdx.x;
    const int lane = t & 31;
    const int wid  = t >> 5;         // 0..7, wave owns q rows [wid*16, wid*16+16)
    const int l15  = lane & 15;
    const int lh   = lane >> 4;
    const int wm   = wid * 16;

    const int bh = blockIdx.y;
    const int b  = bh >> 3;
    const int h  = bh & 7;
    const int q0 = blockIdx.x * 128;

    const bf16_t* qb = qkv + (size_t)b * N_ * QKV_LD + h * DH_;
    const bf16_t* kb = qb + INNER_;
    const bf16_t* vb = qb + 2 * INNER_;

    // --- stage Q tile (128x64): 2 threads/row, 4 async b128 each ---
    {
        const int row = t >> 1, seg = (t & 1) * 32;
        const bf16_t* src = qb + (size_t)(q0 + row) * QKV_LD + seg;
        bf16_t* dst = sQ + row * 72 + seg;
        async_copy_b128(src,      dst);
        async_copy_b128(src + 8,  dst + 8);
        async_copy_b128(src + 16, dst + 16);
        async_copy_b128(src + 24, dst + 24);
    }
    async_wait0();
    __syncthreads();

    // Q A-fragments (16 rows x 64 K) live in registers for the whole kernel.
    FragBF aq[2];
#pragma unroll
    for (int kk = 0; kk < 2; ++kk) {
        const bf16_t* p = sQ + (wm + l15) * 72 + kk * 32 + lh * 8;
        aq[kk].u[0] = *(const uint4*)(p);
        aq[kk].u[1] = *(const uint4*)(p + 16);
    }

    float mrun[8], lrun[8];
    v8f o[4];
    const v8f vzero = {0.f, 0.f, 0.f, 0.f, 0.f, 0.f, 0.f, 0.f};
#pragma unroll
    for (int r = 0; r < 8; ++r) { mrun[r] = -3.0e38f; lrun[r] = 0.f; }
#pragma unroll
    for (int nf = 0; nf < 4; ++nf) o[nf] = vzero;

    const float scale_l2 = 0.125f * LOG2E_;  // fold 1/sqrt(d) into log2 domain
    const int ibase = wm + lh * 8;

    const int krow = t >> 2, kseg = (t & 3) * 16;  // K/V tiles: 4 thr/row, 16 bf16

    for (int kc = 0; kc < 16; ++kc) {
        const int k0g = kc * 64;
        __syncthreads();  // previous chunk's reads of sK/sVt complete
        // stage K tile [key][d] via async copies
        {
            const bf16_t* src = kb + (size_t)(k0g + krow) * QKV_LD + kseg;
            bf16_t* dst = sK + krow * 72 + kseg;
            async_copy_b128(src,     dst);
            async_copy_b128(src + 8, dst + 8);
        }
        // stage V transposed [d][key]: coalesced b128 loads, b16 LDS stores
        {
            union { uint4 u[2]; bf16_t hh[16]; } vst;
            const uint4* vs =
                (const uint4*)(vb + (size_t)(k0g + krow) * QKV_LD + kseg);
            vst.u[0] = vs[0]; vst.u[1] = vs[1];
#pragma unroll
            for (int j = 0; j < 16; ++j) sVt[(kseg + j) * 72 + krow] = vst.hh[j];
        }
        // stage relative-bias diagonals (pre-scaled by log2e)
        if (t < 191) {
            const int rel = (q0 - k0g) + (t - 63) + (N_ - 1);
            sBias[t] = bias_table[(size_t)rel * H_ + h] * LOG2E_;
        }
        async_wait0();
        __syncthreads();

        // S = Q @ K^T  (16x64 per wave)
        v8f s[4];
#pragma unroll
        for (int nf = 0; nf < 4; ++nf) s[nf] = vzero;
#pragma unroll
        for (int kk = 0; kk < 2; ++kk)
#pragma unroll
            for (int nf = 0; nf < 4; ++nf) {
                FragBF bk;
                const bf16_t* p = sK + (nf * 16 + l15) * 72 + kk * 32 + lh * 8;
                bk.u[0] = *(const uint4*)(p);
                bk.u[1] = *(const uint4*)(p + 16);
                s[nf] = __builtin_amdgcn_wmma_f32_16x16x32_bf16(
                    false, aq[kk].v, false, bk.v, (short)0, s[nf], false, false);
            }

        // online softmax in log2 domain; rows r+8*lh -> 16-lane shfl_xor reduce
#pragma unroll
        for (int r = 0; r < 8; ++r) {
            float rmax = -3.0e38f;
#pragma unroll
            for (int nf = 0; nf < 4; ++nf) {
                const int j = nf * 16 + l15;
                const float lg = s[nf][r] * scale_l2 + sBias[ibase + r - j + 63];
                s[nf][r] = lg;
                rmax = fmaxf(rmax, lg);
            }
#pragma unroll
            for (int mm = 1; mm < 16; mm <<= 1)
                rmax = fmaxf(rmax, __shfl_xor(rmax, mm, 32));
            const float mnew  = fmaxf(mrun[r], rmax);
            const float alpha = fast_exp2(mrun[r] - mnew);
            mrun[r] = mnew;
            float rsum = 0.f;
#pragma unroll
            for (int nf = 0; nf < 4; ++nf) {
                const float pv = fast_exp2(s[nf][r] - mnew);
                s[nf][r] = pv;
                rsum += pv;
            }
#pragma unroll
            for (int mm = 1; mm < 16; mm <<= 1)
                rsum += __shfl_xor(rsum, mm, 32);
            lrun[r] = lrun[r] * alpha + rsum;
#pragma unroll
            for (int nf = 0; nf < 4; ++nf) {
                o[nf][r] *= alpha;
                // C-layout -> LDS (per-wave private rows; in-wave LDS ordering)
                sP[(ibase + r) * 72 + nf * 16 + l15] = (bf16_t)s[nf][r];
            }
        }

        // O += P @ V  (16x64 per wave)
#pragma unroll
        for (int kk = 0; kk < 2; ++kk) {
            FragBF ap;
            const bf16_t* pp = sP + (wm + l15) * 72 + kk * 32 + lh * 8;
            ap.u[0] = *(const uint4*)(pp);
            ap.u[1] = *(const uint4*)(pp + 16);
#pragma unroll
            for (int nf = 0; nf < 4; ++nf) {
                FragBF bv;
                const bf16_t* vp = sVt + (nf * 16 + l15) * 72 + kk * 32 + lh * 8;
                bv.u[0] = *(const uint4*)(vp);
                bv.u[1] = *(const uint4*)(vp + 16);
                o[nf] = __builtin_amdgcn_wmma_f32_16x16x32_bf16(
                    false, ap.v, false, bv.v, (short)0, o[nf], false, false);
            }
        }
    }

    // epilogue: normalize and store (merged heads layout [b, n, h*64 + d])
#pragma unroll
    for (int r = 0; r < 8; ++r) {
        const float inv = 1.0f / lrun[r];
        const int n = q0 + ibase + r;
#pragma unroll
        for (int nf = 0; nf < 4; ++nf) {
            const int d = nf * 16 + l15;
            attn_out[(size_t)(b * N_ + n) * INNER_ + h * DH_ + d] =
                (bf16_t)(o[nf][r] * inv);
        }
    }
}

// ---------------------------------------------------------------------------
// host launcher
// ---------------------------------------------------------------------------
extern "C" void kernel_launch(void* const* d_in, const int* in_sizes, int n_in,
                              void* d_out, int out_size, void* d_ws, size_t ws_size,
                              hipStream_t stream) {
    (void)in_sizes; (void)n_in; (void)out_size; (void)ws_size;

    const float* x          = (const float*)d_in[0];
    const float* w_qkv      = (const float*)d_in[1];
    const float* bias_table = (const float*)d_in[2];
    const float* w_out      = (const float*)d_in[3];
    const float* b_out      = (const float*)d_in[4];
    float* out = (float*)d_out;

    const size_t nx    = (size_t)B_ * N_ * DIM_;        // 4,194,304
    const size_t nwqkv = (size_t)DIM_ * 3 * INNER_;     //   786,432
    const size_t nwout = (size_t)INNER_ * DIM_;         //   262,144
    const size_t nqkv  = (size_t)B_ * N_ * 3 * INNER_;  // 12,582,912
    const size_t nattn = (size_t)B_ * N_ * INNER_;      // 4,194,304

    char* ws = (char*)d_ws;
    bf16_t* xb    = (bf16_t*)ws;  ws += nx * 2;
    bf16_t* wqkvb = (bf16_t*)ws;  ws += nwqkv * 2;
    bf16_t* woutb = (bf16_t*)ws;  ws += nwout * 2;
    bf16_t* qkvb  = (bf16_t*)ws;  ws += nqkv * 2;
    bf16_t* attnb = (bf16_t*)ws;  ws += nattn * 2;

    cvt_f32_bf16<<<(int)((nx    + 255) / 256), 256, 0, stream>>>(x, xb, (int)nx);
    cvt_f32_bf16<<<(int)((nwqkv + 255) / 256), 256, 0, stream>>>(w_qkv, wqkvb, (int)nwqkv);
    cvt_f32_bf16<<<(int)((nwout + 255) / 256), 256, 0, stream>>>(w_out, woutb, (int)nwout);

    // qkv = x @ w_qkv  -> bf16 [8192, 1536]
    gemm_bf16<true, false><<<dim3((3 * INNER_) / 128, (B_ * N_) / 128), 256, 0, stream>>>(
        xb, wqkvb, nullptr, (void*)qkvb, B_ * N_, 3 * INNER_, DIM_);

    // flash attention with relative bias -> bf16 [8192, 512]
    flash_attn_relbias<<<dim3(N_ / 128, B_ * H_), 256, 0, stream>>>(
        qkvb, bias_table, attnb);

    // out = attn @ w_out + b_out -> fp32 [8192, 512]
    gemm_bf16<false, true><<<dim3(DIM_ / 128, (B_ * N_) / 128), 256, 0, stream>>>(
        attnb, woutb, b_out, (void*)out, B_ * N_, DIM_, INNER_);
}
